// GRUDecoder_17171279249743
// MI455X (gfx1250) — matrix operand order repaired and checked by reading, not compiled, not run on hardware
//
#include <hip/hip_runtime.h>

// Problem constants (match reference: B, H, V, T)
#define Bsz 8192
#define Hd  1024
#define H3  3072   // 3*H
#define Vv  64
#define Tt  15

typedef __attribute__((ext_vector_type(16))) __bf16 v16bf;
typedef __attribute__((ext_vector_type(8)))  __bf16 v8bf;
typedef __attribute__((ext_vector_type(8)))  float  v8f;

// Build a 16-element bf16 fragment from two contiguous 16-byte loads.
static __device__ __forceinline__ v16bf load_frag16(const __bf16* p0, const __bf16* p1) {
    v8bf lo = *(const v8bf*)p0;
    v8bf hi = *(const v8bf*)p1;
    return __builtin_shufflevector(lo, hi, 0,1,2,3,4,5,6,7,8,9,10,11,12,13,14,15);
}

// ---------------------------------------------------------------------------
// Prep: bf16 weight copies, gi lookup table G[v][3H] = W_ih[:,v]+b_ih,
//       h0 = context (f32 master + bf16 GEMM copy), tok = SOS(0).
// ---------------------------------------------------------------------------
__global__ __launch_bounds__(256) void gru_prep_kernel(
    const float* __restrict__ ctx, const float* __restrict__ W_ih,
    const float* __restrict__ W_hh, const float* __restrict__ b_ih,
    const float* __restrict__ W_out,
    __bf16* __restrict__ Whh_bf, __bf16* __restrict__ Wout_bf,
    float* __restrict__ G, float* __restrict__ hf0, __bf16* __restrict__ hb0,
    int* __restrict__ tok)
{
    const size_t stride = (size_t)gridDim.x * blockDim.x;
    const size_t tid = (size_t)blockIdx.x * blockDim.x + threadIdx.x;
    for (size_t i = tid; i < (size_t)H3 * Hd; i += stride)
        Whh_bf[i] = (__bf16)W_hh[i];
    for (size_t i = tid; i < (size_t)Vv * Hd; i += stride)
        Wout_bf[i] = (__bf16)W_out[i];
    for (size_t i = tid; i < (size_t)Vv * H3; i += stride) {
        int v = (int)(i / H3), j = (int)(i % H3);
        G[i] = W_ih[(size_t)j * Vv + v] + b_ih[j];
    }
    for (size_t i = tid; i < (size_t)Bsz * Hd; i += stride) {
        float c = ctx[i];
        hf0[i] = c;
        hb0[i] = (__bf16)c;
    }
    for (size_t i = tid; i < (size_t)Bsz; i += stride)
        tok[i] = 0;  // SOS
}

// Fused GRU gate epilogue for one 16x16 accumulator triple.
static __device__ __forceinline__ void gru_epilogue(
    v8f cr, v8f cz, v8f cn, int bbase, int j, int half,
    float bhr, float bhz, float bhn,
    const float* __restrict__ G, const int* __restrict__ tok,
    const float* __restrict__ hf, float* __restrict__ hf_out,
    __bf16* __restrict__ hb_out)
{
#pragma unroll
    for (int v = 0; v < 8; ++v) {
        const int b  = bbase + half * 8 + v;
        const int tk = tok[b];
        const float* g = G + (size_t)tk * H3;
        const float ir = g[j], iz = g[Hd + j], inn = g[2 * Hd + j];
        const float ghr = cr[v] + bhr, ghz = cz[v] + bhz, ghn = cn[v] + bhn;
        const float r = 1.0f / (1.0f + __expf(-(ir + ghr)));
        const float z = 1.0f / (1.0f + __expf(-(iz + ghz)));
        const float n = tanhf(inn + r * ghn);
        const float hp = hf[(size_t)b * Hd + j];
        const float hn = (1.0f - z) * n + z * hp;
        hf_out[(size_t)b * Hd + j] = hn;
        hb_out[(size_t)b * Hd + j] = (__bf16)hn;
    }
}

// ---------------------------------------------------------------------------
// GRU step: gh = h @ W_hh^T (bf16 WMMA, f32 acc) for all 3 gates,
// fused gate math + hidden update.
// Wave tile: 32 batch x 16 cols (two A fragments reuse the three B fragments:
// 10 b128 loads per 6 WMMAs).  Block 256 = 8 waves arranged 2(batch)x4(cols):
// 64 batch x 64 cols per block.  grid = (H/64, B/64).
// ---------------------------------------------------------------------------
__global__ __launch_bounds__(256) void gru_step_kernel(
    const __bf16* __restrict__ hb, const float* __restrict__ hf,
    const __bf16* __restrict__ Whh, const float* __restrict__ G,
    const float* __restrict__ b_hh, const int* __restrict__ tok,
    float* __restrict__ hf_out, __bf16* __restrict__ hb_out)
{
    const int lane = threadIdx.x & 31;
    const int wave = threadIdx.x >> 5;
    const int half = lane >> 4;     // K-half selector (A/B fragment layouts)
    const int ln   = lane & 15;     // row (A) / col (B,C) within tile
    const int wy = wave >> 2, wx = wave & 3;
    const int b0 = blockIdx.y * 64 + wy * 32;   // this wave: batch rows b0..b0+31
    const int j0 = blockIdx.x * 64 + wx * 16;

    v8f cr0 = {}, cz0 = {}, cn0 = {};
    v8f cr1 = {}, cz1 = {}, cn1 = {};
    const __bf16* arow0 = hb  + (size_t)(b0 + ln) * Hd;
    const __bf16* arow1 = hb  + (size_t)(b0 + 16 + ln) * Hd;
    const __bf16* rrow  = Whh + (size_t)(j0 + ln) * Hd;            // gate r rows
    const __bf16* zrow  = Whh + (size_t)(j0 + ln + Hd) * Hd;       // gate z rows
    const __bf16* nrow  = Whh + (size_t)(j0 + ln + 2 * Hd) * Hd;   // gate n rows

    for (int kk = 0; kk < Hd; kk += 32) {
        // A 16x32 bf16 frag: lanes 0-15 hold K 0..15, lanes 16-31 K 16..31 (contiguous)
        const int ka = kk + half * 16;
        v16bf a0 = load_frag16(arow0 + ka, arow0 + ka + 8);
        v16bf a1 = load_frag16(arow1 + ka, arow1 + ka + 8);
        // B 32x16 bf16 frag: lanes 0-15 hold K 0..15, lanes 16-31 K 16..31 (contiguous)
        const int kb = kk + half * 16;
        v16bf br = load_frag16(rrow + kb, rrow + kb + 8);
        v16bf bz = load_frag16(zrow + kb, zrow + kb + 8);
        v16bf bn = load_frag16(nrow + kb, nrow + kb + 8);
        if (kk + 32 < Hd) {
            __builtin_prefetch(arow0 + ka + 32, 0, 3);
            __builtin_prefetch(arow1 + ka + 32, 0, 3);
        }
        cr0 = __builtin_amdgcn_wmma_f32_16x16x32_bf16(false, a0, false, br, (short)0, cr0, false, false);
        cz0 = __builtin_amdgcn_wmma_f32_16x16x32_bf16(false, a0, false, bz, (short)0, cz0, false, false);
        cn0 = __builtin_amdgcn_wmma_f32_16x16x32_bf16(false, a0, false, bn, (short)0, cn0, false, false);
        cr1 = __builtin_amdgcn_wmma_f32_16x16x32_bf16(false, a1, false, br, (short)0, cr1, false, false);
        cz1 = __builtin_amdgcn_wmma_f32_16x16x32_bf16(false, a1, false, bz, (short)0, cz1, false, false);
        cn1 = __builtin_amdgcn_wmma_f32_16x16x32_bf16(false, a1, false, bn, (short)0, cn1, false, false);
    }

    // C/D layout: VGPR v, lanes 0-15 -> (M=v, N=lane), lanes 16-31 -> (M=v+8, N=lane-16)
    const int j = j0 + ln;
    const float bhr = b_hh[j], bhz = b_hh[Hd + j], bhn = b_hh[2 * Hd + j];
    gru_epilogue(cr0, cz0, cn0, b0,      j, half, bhr, bhz, bhn, G, tok, hf, hf_out, hb_out);
    gru_epilogue(cr1, cz1, cn1, b0 + 16, j, half, bhr, bhz, bhn, G, tok, hf, hf_out, hb_out);
}

// ---------------------------------------------------------------------------
// Output step: logits = h @ W_out^T + b_out (16x64 per wave via 4 WMMA accs),
// log-softmax + argmax over V=64 via shfl_xor within 16-lane half,
// write logp[b][t][:], preds[b][t], and feedback token.
// grid = B/(16*8) blocks of 256 threads.
// ---------------------------------------------------------------------------
__global__ __launch_bounds__(256) void gru_out_kernel(
    const __bf16* __restrict__ hb, const __bf16* __restrict__ Wout,
    const float* __restrict__ b_out, float* __restrict__ logp,
    float* __restrict__ preds, int* __restrict__ tok, int t)
{
    const int lane = threadIdx.x & 31;
    const int wave = threadIdx.x >> 5;
    const int half = lane >> 4;
    const int ln   = lane & 15;
    const int b0 = (blockIdx.x * 8 + wave) * 16;

    v8f a0 = {}, a1 = {}, a2 = {}, a3 = {};
    const __bf16* arow = hb + (size_t)(b0 + ln) * Hd;
    const __bf16* w0 = Wout + (size_t)(ln) * Hd;
    const __bf16* w1 = Wout + (size_t)(16 + ln) * Hd;
    const __bf16* w2 = Wout + (size_t)(32 + ln) * Hd;
    const __bf16* w3 = Wout + (size_t)(48 + ln) * Hd;

    for (int kk = 0; kk < Hd; kk += 32) {
        v16bf a = load_frag16(arow + kk + half * 16, arow + kk + half * 16 + 8);
        const int kb = kk + half * 16;
        v16bf bv0 = load_frag16(w0 + kb, w0 + kb + 8);
        v16bf bv1 = load_frag16(w1 + kb, w1 + kb + 8);
        v16bf bv2 = load_frag16(w2 + kb, w2 + kb + 8);
        v16bf bv3 = load_frag16(w3 + kb, w3 + kb + 8);
        a0 = __builtin_amdgcn_wmma_f32_16x16x32_bf16(false, a, false, bv0, (short)0, a0, false, false);
        a1 = __builtin_amdgcn_wmma_f32_16x16x32_bf16(false, a, false, bv1, (short)0, a1, false, false);
        a2 = __builtin_amdgcn_wmma_f32_16x16x32_bf16(false, a, false, bv2, (short)0, a2, false, false);
        a3 = __builtin_amdgcn_wmma_f32_16x16x32_bf16(false, a, false, bv3, (short)0, a3, false, false);
    }

    const float bo0 = b_out[ln], bo1 = b_out[16 + ln], bo2 = b_out[32 + ln], bo3 = b_out[48 + ln];

#pragma unroll
    for (int v = 0; v < 8; ++v) {
        const int b = b0 + half * 8 + v;
        float lv0 = a0[v] + bo0, lv1 = a1[v] + bo1, lv2 = a2[v] + bo2, lv3 = a3[v] + bo3;

        // max over the 64 vocab entries (4 local + 16 lanes within the half)
        float m = fmaxf(fmaxf(lv0, lv1), fmaxf(lv2, lv3));
        for (int msk = 1; msk <= 8; msk <<= 1)
            m = fmaxf(m, __shfl_xor(m, msk));
        // sum of exp
        float s = __expf(lv0 - m) + __expf(lv1 - m) + __expf(lv2 - m) + __expf(lv3 - m);
        for (int msk = 1; msk <= 8; msk <<= 1)
            s += __shfl_xor(s, msk);
        const float lse = m + __logf(s);

        const size_t ob = ((size_t)b * Tt + t) * Vv;
        logp[ob +      ln] = lv0 - lse;
        logp[ob + 16 + ln] = lv1 - lse;
        logp[ob + 32 + ln] = lv2 - lse;
        logp[ob + 48 + ln] = lv3 - lse;

        // argmax (first-max tie break: smaller index wins)
        float bvv = lv0; int bii = ln;
        if (lv1 > bvv) { bvv = lv1; bii = 16 + ln; }
        if (lv2 > bvv) { bvv = lv2; bii = 32 + ln; }
        if (lv3 > bvv) { bvv = lv3; bii = 48 + ln; }
        for (int msk = 1; msk <= 8; msk <<= 1) {
            float ov = __shfl_xor(bvv, msk);
            int   oi = __shfl_xor(bii, msk);
            if (ov > bvv || (ov == bvv && oi < bii)) { bvv = ov; bii = oi; }
        }
        if (ln == 0) {
            preds[(size_t)b * Tt + t] = (float)bii;
            tok[b] = bii;
        }
    }
}

// ---------------------------------------------------------------------------
extern "C" void kernel_launch(void* const* d_in, const int* in_sizes, int n_in,
                              void* d_out, int out_size, void* d_ws, size_t ws_size,
                              hipStream_t stream) {
    const float* ctx   = (const float*)d_in[0];
    const float* W_ih  = (const float*)d_in[1];
    const float* W_hh  = (const float*)d_in[2];
    const float* b_ih  = (const float*)d_in[3];
    const float* b_hh  = (const float*)d_in[4];
    const float* W_out = (const float*)d_in[5];
    const float* b_out = (const float*)d_in[6];
    (void)in_sizes; (void)n_in; (void)out_size; (void)ws_size;

    char* ws = (char*)d_ws;
    size_t off = 0;
    auto take = [&](size_t bytes) -> void* {
        void* p = ws + off;
        off += (bytes + 255) & ~(size_t)255;
        return p;
    };
    __bf16* Whh_bf  = (__bf16*)take((size_t)H3 * Hd * sizeof(__bf16));
    __bf16* Wout_bf = (__bf16*)take((size_t)Vv * Hd * sizeof(__bf16));
    float*  G       = (float*) take((size_t)Vv * H3 * sizeof(float));
    float*  hf[2];
    __bf16* hbuf[2];
    hf[0]   = (float*) take((size_t)Bsz * Hd * sizeof(float));
    hf[1]   = (float*) take((size_t)Bsz * Hd * sizeof(float));
    hbuf[0] = (__bf16*)take((size_t)Bsz * Hd * sizeof(__bf16));
    hbuf[1] = (__bf16*)take((size_t)Bsz * Hd * sizeof(__bf16));
    int* tok = (int*)take((size_t)Bsz * sizeof(int));

    float* logp  = (float*)d_out;                                  // [B, T, V]
    float* preds = (float*)d_out + (size_t)Bsz * Tt * Vv;          // [B, T]

    gru_prep_kernel<<<1024, 256, 0, stream>>>(
        ctx, W_ih, W_hh, b_ih, W_out, Whh_bf, Wout_bf, G, hf[0], hbuf[0], tok);

    dim3 g1(Hd / 64, Bsz / 64);          // (16, 128)
    const int g2 = Bsz / (16 * 8);       // 64
    for (int t = 0; t < Tt; ++t) {
        const int src = t & 1, dst = src ^ 1;
        gru_step_kernel<<<g1, 256, 0, stream>>>(
            hbuf[src], hf[src], Whh_bf, G, b_hh, tok, hf[dst], hbuf[dst]);
        gru_out_kernel<<<g2, 256, 0, stream>>>(
            hbuf[dst], Wout_bf, b_out, logp, preds, tok, t);
    }
}
